// SelfAttention_30099130810750
// MI455X (gfx1250) — matrix operand order repaired and checked by reading, not compiled, not run on hardware
//
#include <hip/hip_runtime.h>

// ---------------------------------------------------------------------------
// Self-attention (softmax over query axis) for MI455X / gfx1250, wave32 WMMA.
// E=64, H=8, B=4, T=2048.  All GEMMs via v_wmma_f32_16x16x32_bf16.
// ---------------------------------------------------------------------------

#define E_DIM 64
#define H_DIM 8
#define B_DIM 4
#define T_DIM 2048
#define BT    (B_DIM * T_DIM)      // 8192
#define NOUT  (H_DIM * E_DIM)      // 512

typedef __bf16 bf16_t;
typedef __attribute__((ext_vector_type(16))) __bf16 v16bf;
typedef __attribute__((ext_vector_type(8)))  __bf16 v8bf;
typedef __attribute__((ext_vector_type(8)))  float  v8f;

// WMMA helper: D = A(16x32 bf16) * B(32x16 bf16) + C(16x16 f32)
__device__ __forceinline__ v8f wmma_bf16(v16bf a, v16bf b, v8f c) {
    return __builtin_amdgcn_wmma_f32_16x16x32_bf16(
        /*neg_a=*/false, a, /*neg_b=*/false, b,
        /*c_mod=*/(short)0, c, /*reuse_a=*/false, /*reuse_b=*/false);
}

// A-operand lane load from bf16 row-major source:
// lane holds row = lane%16, 16 contiguous elems at kbase.
__device__ __forceinline__ v16bf load_a_bf16(const bf16_t* p) {
    v8bf lo = *(const v8bf*)(p);
    v8bf hh = *(const v8bf*)(p + 8);
    return __builtin_shufflevector(lo, hh, 0,1,2,3,4,5,6,7,8,9,10,11,12,13,14,15);
}

// ---------------------------------------------------------------------------
// Kernel 0: weight transpose + fp32->bf16 convert.
// WvT/WkT/WqT: [512][64] with WT[n][k] = W[k][n];  WuT: [64][512].
// ---------------------------------------------------------------------------
__global__ void prep_weights(const float* __restrict__ Wv, const float* __restrict__ Wk,
                             const float* __restrict__ Wq, const float* __restrict__ Wu,
                             bf16_t* __restrict__ WvT, bf16_t* __restrict__ WkT,
                             bf16_t* __restrict__ WqT, bf16_t* __restrict__ WuT) {
    int idx   = blockIdx.x * 256 + threadIdx.x;     // 0 .. 131071
    int which = idx >> 15;                          // 0..3
    int r     = idx & 32767;
    if (which < 3) {
        const float* W  = (which == 0) ? Wv  : (which == 1) ? Wk  : Wq;
        bf16_t*      WT = (which == 0) ? WvT : (which == 1) ? WkT : WqT;
        int n = r >> 6, k = r & 63;                 // n: out-col [0,512), k: [0,64)
        WT[n * 64 + k] = (bf16_t)W[k * NOUT + n];
    } else {
        int n = r >> 9, k = r & 511;                // n: [0,64), k: [0,512)
        WuT[n * NOUT + k] = (bf16_t)Wu[k * E_DIM + n];
    }
}

// ---------------------------------------------------------------------------
// Kernel 1: projection  out = X[BT,64] @ W[64,512]  (WMMA, K=64 in 2 steps).
// TRANSPOSED==0: store out[((b*H+h)*T + t)*64 + e]   (Q, K layout)
// TRANSPOSED==1: store out[((b*H+h)*64 + e)*T + t]   (V^T layout)
// grid (BT/16, 4), block 256 (8 waves, 16 N-cols each).
// ---------------------------------------------------------------------------
template <int TRANSPOSED>
__global__ void proj_kernel(const float* __restrict__ X, const bf16_t* __restrict__ WT,
                            bf16_t* __restrict__ out) {
    const int lane = threadIdx.x & 31;
    const int wave = threadIdx.x >> 5;
    const int hi   = lane >> 4;
    const int lr   = lane & 15;
    const int mtile = blockIdx.x;
    const int ntile = blockIdx.y * 8 + wave;
    const int arow  = mtile * 16 + lr;

    v8f c = {};
#pragma unroll
    for (int ks = 0; ks < 2; ++ks) {
        // A operand: fp32 -> bf16 on the fly
        const float* pa = X + (size_t)arow * E_DIM + ks * 32 + hi * 16;
        v16bf a;
#pragma unroll
        for (int i = 0; i < 8; ++i) {
            a[i]     = (bf16_t)pa[i];
            a[8 + i] = (bf16_t)pa[8 + i];
        }
        // B operand: 16 contiguous bf16 from transposed weights
        v16bf bm = *(const v16bf*)(WT + (size_t)(ntile * 16 + lr) * E_DIM + ks * 32 + hi * 16);
        c = wmma_bf16(a, bm, c);
    }
    const int colg = ntile * 16 + lr;
    const int h = colg >> 6, e = colg & 63;
#pragma unroll
    for (int r = 0; r < 8; ++r) {
        int mg = mtile * 16 + r + 8 * hi;           // global row in [0, BT)
        int b  = mg >> 11;                          // / T
        int t  = mg & (T_DIM - 1);
        size_t dst;
        if (TRANSPOSED)
            dst = ((size_t)((b * H_DIM + h) * E_DIM + e)) * T_DIM + t;
        else
            dst = ((size_t)((b * H_DIM + h) * T_DIM + t)) * E_DIM + e;
        out[dst] = (bf16_t)c[r];
    }
}

// ---------------------------------------------------------------------------
// Kernel 2: column softmax stats over the QUERY axis.
// For each (b,h) and key column d: m_d = max_t S[t,d], Rz_d = 1/sum_t exp(S-m).
// grid (B*H, T/256), block 256; each wave owns 32 key columns, streams all t
// (the Q A-operand load is shared between the two column tiles).
// ---------------------------------------------------------------------------
__global__ void colstats_kernel(const bf16_t* __restrict__ Qb, const bf16_t* __restrict__ Kb,
                                float* __restrict__ Mcol, float* __restrict__ Rz) {
    const int lane = threadIdx.x & 31;
    const int wave = threadIdx.x >> 5;
    const int hi   = lane >> 4;
    const int lr   = lane & 15;
    const int bh    = blockIdx.x;
    const int dbase = blockIdx.y * 256 + wave * 32;
    const bf16_t* Qh = Qb + (size_t)bh * T_DIM * E_DIM;
    const bf16_t* Kh = Kb + (size_t)bh * T_DIM * E_DIM;

    // Loop-invariant B operands for two 16-column tiles: B[e][d] = K[d][e]
    const bf16_t* kp0 = Kh + (size_t)(dbase + lr) * E_DIM + hi * 16;
    const bf16_t* kp1 = Kh + (size_t)(dbase + 16 + lr) * E_DIM + hi * 16;
    v16bf b0 = *(const v16bf*)(kp0);
    v16bf b1 = *(const v16bf*)(kp0 + 32);
    v16bf b2 = *(const v16bf*)(kp1);
    v16bf b3 = *(const v16bf*)(kp1 + 32);

    const float scale = 0.125f;                      // 1/sqrt(64)
    float m0 = -3.0e38f, z0 = 0.0f;
    float m1 = -3.0e38f, z1 = 0.0f;
    for (int tt = 0; tt < T_DIM; tt += 16) {
        const bf16_t* qp = Qh + (size_t)(tt + lr) * E_DIM + hi * 16;
        __builtin_prefetch(Qh + (size_t)(((tt + 64) & (T_DIM - 1)) + lr) * E_DIM, 0, 3);
        v16bf a0 = load_a_bf16(qp);
        v16bf a1 = load_a_bf16(qp + 32);
        v8f c0 = {}, c1 = {};
        c0 = wmma_bf16(a0, b0, c0);
        c0 = wmma_bf16(a1, b1, c0);
        c1 = wmma_bf16(a0, b2, c1);
        c1 = wmma_bf16(a1, b3, c1);
        float tm0 = -3.0e38f, tm1 = -3.0e38f;
#pragma unroll
        for (int r = 0; r < 8; ++r) {
            tm0 = fmaxf(tm0, c0[r] * scale);
            tm1 = fmaxf(tm1, c1[r] * scale);
        }
        float nm0 = fmaxf(m0, tm0), nm1 = fmaxf(m1, tm1);
        float zs0 = 0.0f, zs1 = 0.0f;
#pragma unroll
        for (int r = 0; r < 8; ++r) {
            zs0 += __expf(c0[r] * scale - nm0);
            zs1 += __expf(c1[r] * scale - nm1);
        }
        z0 = z0 * __expf(m0 - nm0) + zs0;  m0 = nm0;
        z1 = z1 * __expf(m1 - nm1) + zs1;  m1 = nm1;
    }
    // combine the two lane halves (rows 0-7 vs 8-15 of each column)
    float om0 = __shfl_xor(m0, 16, 32), oz0 = __shfl_xor(z0, 16, 32);
    float om1 = __shfl_xor(m1, 16, 32), oz1 = __shfl_xor(z1, 16, 32);
    float nm0 = fmaxf(m0, om0);
    float nm1 = fmaxf(m1, om1);
    float zz0 = z0 * __expf(m0 - nm0) + oz0 * __expf(om0 - nm0);
    float zz1 = z1 * __expf(m1 - nm1) + oz1 * __expf(om1 - nm1);
    if (hi == 0) {
        Mcol[(size_t)bh * T_DIM + dbase + lr]      = nm0;
        Rz[(size_t)bh * T_DIM + dbase + lr]        = 1.0f / zz0;
        Mcol[(size_t)bh * T_DIM + dbase + 16 + lr] = nm1;
        Rz[(size_t)bh * T_DIM + dbase + 16 + lr]   = 1.0f / zz1;
    }
}

// ---------------------------------------------------------------------------
// Kernel 3: out[t,:] = sum_d exp(S[t,d]-m_d)/Z_d * V[d,:]   (per b,h)
// Each wave owns a 32-query tile (two 16-row A tiles sharing every K/V
// B-operand load) and all 64 output cols; streams keys in chunks of 32.
// P tiles bounced through wave-private LDS (in-order per wave -> no barrier)
// to convert C-layout -> A-operand layout.
// grid (B*H, T/256), block 256, 16 KB LDS.
// ---------------------------------------------------------------------------
__global__ void attnout_kernel(const bf16_t* __restrict__ Qb, const bf16_t* __restrict__ Kb,
                               const bf16_t* __restrict__ Vt, const float* __restrict__ Mcol,
                               const float* __restrict__ Rz, bf16_t* __restrict__ AO) {
    __shared__ __align__(16) bf16_t stage[8 * 32 * 32];   // 16 KB, one 32x32 tile per wave
    const int lane = threadIdx.x & 31;
    const int wave = threadIdx.x >> 5;
    const int hi   = lane >> 4;
    const int lr   = lane & 15;
    const int bh    = blockIdx.x;
    const int b     = bh / H_DIM, h = bh % H_DIM;
    const int trow0 = (blockIdx.y * 8 + wave) * 32;       // 32 queries per wave
    const bf16_t* Qh = Qb + (size_t)bh * T_DIM * E_DIM;
    const bf16_t* Kh = Kb + (size_t)bh * T_DIM * E_DIM;
    const bf16_t* Vh = Vt + (size_t)bh * E_DIM * T_DIM;
    const float*  mc = Mcol + (size_t)bh * T_DIM;
    const float*  rz = Rz   + (size_t)bh * T_DIM;
    bf16_t* st = stage + wave * 1024;                     // [32 t][32 d] bf16

    // Loop-invariant Q A-operands for the two 16-query tiles
    const bf16_t* qp0 = Qh + (size_t)(trow0 + lr) * E_DIM + hi * 16;
    const bf16_t* qp1 = Qh + (size_t)(trow0 + 16 + lr) * E_DIM + hi * 16;
    v16bf a0 = load_a_bf16(qp0);
    v16bf a1 = load_a_bf16(qp0 + 32);
    v16bf a2 = load_a_bf16(qp1);
    v16bf a3 = load_a_bf16(qp1 + 32);

    const float scale = 0.125f;
    v8f acc0[4] = {};
    v8f acc1[4] = {};
    for (int d = 0; d < T_DIM; d += 32) {
        __builtin_prefetch(Kh + (size_t)(((d + 64) & (T_DIM - 1)) + lr) * E_DIM, 0, 3);
#pragma unroll
        for (int dd = 0; dd < 32; dd += 16) {
            const int dcol = d + dd + lr;
            const bf16_t* kp = Kh + (size_t)dcol * E_DIM + hi * 16;
            v16bf b0 = *(const v16bf*)(kp);
            v16bf b1 = *(const v16bf*)(kp + 32);
            v8f c0 = {}, c1 = {};
            c0 = wmma_bf16(a0, b0, c0);
            c0 = wmma_bf16(a1, b1, c0);
            c1 = wmma_bf16(a2, b0, c1);
            c1 = wmma_bf16(a3, b1, c1);
            const float mcol = mc[dcol];
            const float rzc  = rz[dcol];
#pragma unroll
            for (int r = 0; r < 8; ++r) {
                float p0 = __expf(c0[r] * scale - mcol) * rzc;
                float p1 = __expf(c1[r] * scale - mcol) * rzc;
                st[(r + 8 * hi) * 32 + dd + lr]        = (bf16_t)p0;  // t_local 0..15
                st[(16 + r + 8 * hi) * 32 + dd + lr]   = (bf16_t)p1;  // t_local 16..31
            }
        }
        // Re-read P in A-operand layout (same wave: LDS ops are in-order)
        v16bf pa0 = load_a_bf16(st + lr * 32 + hi * 16);
        v16bf pa1 = load_a_bf16(st + (16 + lr) * 32 + hi * 16);
        // 4 output-column tiles; B[d][e'] = Vt[e'][d] -> contiguous loads
#pragma unroll
        for (int et = 0; et < 4; ++et) {
            v16bf bv = *(const v16bf*)(Vh + (size_t)(et * 16 + lr) * T_DIM + d + hi * 16);
            acc0[et] = wmma_bf16(pa0, bv, acc0[et]);
            acc1[et] = wmma_bf16(pa1, bv, acc1[et]);
        }
    }
    // Store attention output (A-operand layout for unify): AO[b*T+t][h*64+e]
#pragma unroll
    for (int et = 0; et < 4; ++et) {
#pragma unroll
        for (int r = 0; r < 8; ++r) {
            size_t row0 = (size_t)(b * T_DIM + trow0 + r + 8 * hi);
            AO[row0 * NOUT + h * E_DIM + et * 16 + lr]        = (bf16_t)acc0[et][r];
            AO[(row0 + 16) * NOUT + h * E_DIM + et * 16 + lr] = (bf16_t)acc1[et][r];
        }
    }
}

// ---------------------------------------------------------------------------
// Kernel 4: unify  out = AO[BT,512] @ Wu[512,64] + bu   (fp32 out)
// grid (BT/32), block 256: 8 waves = 2 M-tiles x 4 N-tiles.
// ---------------------------------------------------------------------------
__global__ void unify_kernel(const bf16_t* __restrict__ AO, const bf16_t* __restrict__ WuT,
                             const float* __restrict__ bu, float* __restrict__ out) {
    const int lane = threadIdx.x & 31;
    const int wave = threadIdx.x >> 5;
    const int hi   = lane >> 4;
    const int lr   = lane & 15;
    const int mtile = blockIdx.x * 2 + (wave >> 2);
    const int ntile = wave & 3;
    const int arow  = mtile * 16 + lr;

    v8f c = {};
#pragma unroll
    for (int ks = 0; ks < 16; ++ks) {
        v16bf a  = load_a_bf16(AO + (size_t)arow * NOUT + ks * 32 + hi * 16);
        v16bf bm = *(const v16bf*)(WuT + (size_t)(ntile * 16 + lr) * NOUT + ks * 32 + hi * 16);
        c = wmma_bf16(a, bm, c);
    }
    const int col  = ntile * 16 + lr;
    const float bias = bu[col];
#pragma unroll
    for (int r = 0; r < 8; ++r) {
        out[(size_t)(mtile * 16 + r + 8 * hi) * E_DIM + col] = c[r] + bias;
    }
}

// ---------------------------------------------------------------------------
// Host launcher
// ---------------------------------------------------------------------------
extern "C" void kernel_launch(void* const* d_in, const int* in_sizes, int n_in,
                              void* d_out, int out_size, void* d_ws, size_t ws_size,
                              hipStream_t stream) {
    const float* values  = (const float*)d_in[0];
    const float* keys    = (const float*)d_in[1];
    const float* queries = (const float*)d_in[2];
    const float* Wv      = (const float*)d_in[3];
    const float* Wk      = (const float*)d_in[4];
    const float* Wq      = (const float*)d_in[5];
    const float* Wu      = (const float*)d_in[6];
    const float* bu      = (const float*)d_in[7];

    char* ws = (char*)d_ws;
    size_t off = 0;
    auto carve = [&](size_t bytes) -> void* {
        void* p = ws + off;
        off += (bytes + 255) & ~(size_t)255;
        return p;
    };
    bf16_t* WvT = (bf16_t*)carve(NOUT * E_DIM * sizeof(bf16_t));       // 64 KB
    bf16_t* WkT = (bf16_t*)carve(NOUT * E_DIM * sizeof(bf16_t));
    bf16_t* WqT = (bf16_t*)carve(NOUT * E_DIM * sizeof(bf16_t));
    bf16_t* WuT = (bf16_t*)carve(E_DIM * NOUT * sizeof(bf16_t));
    bf16_t* Qb  = (bf16_t*)carve((size_t)BT * E_DIM * H_DIM * sizeof(bf16_t)); // 8 MB
    bf16_t* Kb  = (bf16_t*)carve((size_t)BT * E_DIM * H_DIM * sizeof(bf16_t));
    bf16_t* Vt  = (bf16_t*)carve((size_t)BT * E_DIM * H_DIM * sizeof(bf16_t));
    float*  Mc  = (float*)carve((size_t)B_DIM * H_DIM * T_DIM * sizeof(float));
    float*  Rz  = (float*)carve((size_t)B_DIM * H_DIM * T_DIM * sizeof(float));
    bf16_t* AO  = (bf16_t*)carve((size_t)BT * NOUT * sizeof(bf16_t));  // 8 MB
    (void)ws_size; (void)in_sizes; (void)n_in; (void)out_size;

    prep_weights<<<512, 256, 0, stream>>>(Wv, Wk, Wq, Wu, WvT, WkT, WqT, WuT);

    dim3 gproj(BT / 16, 4);
    proj_kernel<0><<<gproj, 256, 0, stream>>>(queries, WqT, Qb);
    proj_kernel<0><<<gproj, 256, 0, stream>>>(keys,    WkT, Kb);
    proj_kernel<1><<<gproj, 256, 0, stream>>>(values,  WvT, Vt);

    dim3 gattn(B_DIM * H_DIM, T_DIM / 256);
    colstats_kernel<<<gattn, 256, 0, stream>>>(Qb, Kb, Mc, Rz);
    attnout_kernel<<<gattn, 256, 0, stream>>>(Qb, Kb, Vt, Mc, Rz, AO);

    unify_kernel<<<BT / 32, 256, 0, stream>>>(AO, WuT, bu, (float*)d_out);
}
